// MoIETransformerBlock_26817775796690
// MI455X (gfx1250) — matrix operand order repaired and checked by reading, not compiled, not run on hardware
//
#include <hip/hip_runtime.h>
#include <cstdint>
#include <cstddef>

// ---------------------------------------------------------------------------
// MI455X (gfx1250) MoIE transformer block.
// All heavy GEMMs use v_wmma_f32_16x16x32_bf16 (wave32 WMMA), fp32 accumulate.
// ---------------------------------------------------------------------------

#define DEV __device__ __forceinline__

typedef __attribute__((ext_vector_type(16))) __bf16 v16bf;
typedef __attribute__((ext_vector_type(8)))  __bf16 v8bf;
typedef __attribute__((ext_vector_type(8)))  float  v8f;

namespace cfg {
constexpr int B   = 8;
constexpr int S   = 1024;
constexpr int D   = 1024;
constexpr int H   = 16;
constexpr int HD  = 64;
constexpr int DFF = 4096;
constexpr int N   = B * S;      // 8192 tokens
constexpr int D3  = 3 * D;      // 3072
constexpr float EPS_COS = 1e-8f;
constexpr float EPS_LN  = 1e-5f;
}

DEV float  tof(__bf16 x) { return (float)x; }
DEV __bf16 tob(float x)  { return (__bf16)x; }

DEV v8f zero8() {
  v8f z;
#pragma unroll
  for (int i = 0; i < 8; ++i) z[i] = 0.f;
  return z;
}

// A-matrix fragment (16x32 bf16, MxK). Lane: m = lane&15, kh = lane>>4.
// Element e -> k = k0 + kh*16 + e  (16 contiguous along K of row m)
DEV v16bf loadA(const __bf16* __restrict__ rowPtr, int k0, int kh) {
  return *(const v16bf*)(rowPtr + k0 + kh * 16);
}

// B-matrix fragment (32x16 bf16, KxN). Lane: n = lane&15, kh = lane>>4.
// Element e -> k = k0 + kh*16 + e  (16 contiguous along K of column n)
DEV v16bf loadB(const __bf16* __restrict__ colPtr, int k0, int kh) {
  return *(const v16bf*)(colPtr + k0 + kh * 16);
}

DEV v8f wmma_bf16(v16bf a, v16bf b, v8f c) {
  return __builtin_amdgcn_wmma_f32_16x16x32_bf16(false, a, false, b,
                                                 (short)0, c, false, false);
}

// ---------------------------------------------------------------------------
// fp32 -> bf16 elementwise conversion (weights)
// ---------------------------------------------------------------------------
__global__ void k_cvt_bf16(const float* __restrict__ src,
                           __bf16* __restrict__ dst, int n) {
  int i = blockIdx.x * blockDim.x + threadIdx.x;
  int stride = gridDim.x * blockDim.x;
  for (; i < n; i += stride) dst[i] = tob(src[i]);
}

// ---------------------------------------------------------------------------
// keys = mu * softplus(sigma); kn = max(||keys_row||, eps); also mu -> bf16
// one block (256 thr) per DFF row
// ---------------------------------------------------------------------------
__global__ void k_keys(const float* __restrict__ mu,
                       const float* __restrict__ sigma,
                       __bf16* __restrict__ keys_bf,
                       __bf16* __restrict__ mu_bf,
                       float* __restrict__ kn) {
  __shared__ float red[256];
  const int r = blockIdx.x, t = threadIdx.x;
  const float* mrow = mu + (size_t)r * cfg::D;
  const float* srow = sigma + (size_t)r * cfg::D;
  float ss = 0.f;
#pragma unroll
  for (int j = 0; j < 4; ++j) {
    int i = j * 256 + t;
    float s  = srow[i];
    float sp = (s > 20.f) ? s : log1pf(__expf(s));
    float kv = mrow[i] * sp;
    keys_bf[(size_t)r * cfg::D + i] = tob(kv);
    mu_bf[(size_t)r * cfg::D + i]   = tob(mrow[i]);
    ss += kv * kv;
  }
  red[t] = ss; __syncthreads();
  for (int o = 128; o > 0; o >>= 1) {
    if (t < o) red[t] += red[t + o];
    __syncthreads();
  }
  if (t == 0) kn[r] = fmaxf(sqrtf(red[0]), cfg::EPS_COS);
}

// ---------------------------------------------------------------------------
// LayerNorm: one block (256 thr) per row of D=1024; bf16 out (+optional norm)
// ---------------------------------------------------------------------------
__global__ void k_layernorm(const float* __restrict__ X,
                            const float* __restrict__ w,
                            const float* __restrict__ b,
                            __bf16* __restrict__ Y,
                            float* __restrict__ rown) {
  __shared__ float red[256];
  const int r = blockIdx.x, t = threadIdx.x;
  const float* xr = X + (size_t)r * cfg::D;
  float x[4];
  float s = 0.f;
#pragma unroll
  for (int j = 0; j < 4; ++j) { x[j] = xr[j * 256 + t]; s += x[j]; }
  red[t] = s; __syncthreads();
  for (int o = 128; o > 0; o >>= 1) { if (t < o) red[t] += red[t + o]; __syncthreads(); }
  const float mean = red[0] * (1.f / cfg::D);
  __syncthreads();
  float v = 0.f;
#pragma unroll
  for (int j = 0; j < 4; ++j) { float d = x[j] - mean; v += d * d; }
  red[t] = v; __syncthreads();
  for (int o = 128; o > 0; o >>= 1) { if (t < o) red[t] += red[t + o]; __syncthreads(); }
  const float rs = rsqrtf(red[0] * (1.f / cfg::D) + cfg::EPS_LN);
  float nn = 0.f;
#pragma unroll
  for (int j = 0; j < 4; ++j) {
    int i = j * 256 + t;
    float y = (x[j] - mean) * rs * w[i] + b[i];
    Y[(size_t)r * cfg::D + i] = tob(y);
    nn += y * y;
  }
  if (rown) {
    __syncthreads();
    red[t] = nn; __syncthreads();
    for (int o = 128; o > 0; o >>= 1) { if (t < o) red[t] += red[t + o]; __syncthreads(); }
    if (t == 0) rown[r] = fmaxf(sqrtf(red[0]), cfg::EPS_COS);
  }
}

// ---------------------------------------------------------------------------
// Generic WMMA GEMM: C[M,Nout] = A[M,K](bf16) @ W[Nout,K]^T(bf16) + bias
// Each wave computes a 32x32 tile (2x2 of 16x16), K stepped by 32.
// outb != null : store bf16        (QKV)
// outb == null : outf = res + C    (out_proj / FFN2 residual epilogue)
// ---------------------------------------------------------------------------
__global__ void k_gemm_bf16(const __bf16* __restrict__ A,
                            const __bf16* __restrict__ W,
                            const float* __restrict__ bias,
                            const float* __restrict__ res,
                            float* __restrict__ outf,
                            __bf16* __restrict__ outb,
                            int M, int Nout, int K) {
  const int lane = threadIdx.x & 31;
  const int kh   = lane >> 4;
  const int ln16 = lane & 15;
  const int gtile = blockIdx.x * (blockDim.x >> 5) + (threadIdx.x >> 5);
  const int tilesN = Nout >> 5;
  const int tm = gtile / tilesN, tn = gtile % tilesN;
  const int r0 = tm * 32, c0 = tn * 32;

  const __bf16* arow0 = A + (size_t)(r0 + ln16) * K;
  const __bf16* arow1 = arow0 + (size_t)16 * K;
  const __bf16* wcol0 = W + (size_t)(c0 + ln16) * K;
  const __bf16* wcol1 = wcol0 + (size_t)16 * K;

  v8f acc[2][2];
#pragma unroll
  for (int i = 0; i < 2; ++i)
#pragma unroll
    for (int j = 0; j < 2; ++j) acc[i][j] = zero8();

  for (int k0 = 0; k0 < K; k0 += 32) {
    __builtin_prefetch(arow0 + k0 + 128, 0, 1);   // global_prefetch next A
    v16bf a0 = loadA(arow0, k0, kh);
    v16bf a1 = loadA(arow1, k0, kh);
    v16bf b0 = loadB(wcol0, k0, kh);
    v16bf b1 = loadB(wcol1, k0, kh);
    acc[0][0] = wmma_bf16(a0, b0, acc[0][0]);
    acc[0][1] = wmma_bf16(a0, b1, acc[0][1]);
    acc[1][0] = wmma_bf16(a1, b0, acc[1][0]);
    acc[1][1] = wmma_bf16(a1, b1, acc[1][1]);
  }

#pragma unroll
  for (int i = 0; i < 2; ++i) {
#pragma unroll
    for (int j = 0; j < 2; ++j) {
      const int col = c0 + 16 * j + ln16;
      const float bv = bias ? bias[col] : 0.f;
#pragma unroll
      for (int v = 0; v < 8; ++v) {
        const int row = r0 + 16 * i + v + 8 * kh;   // C/D layout: M = v + 8*(lane>=16)
        const size_t idx = (size_t)row * Nout + col;
        const float val = acc[i][j][v] + bv;
        if (outb) outb[idx] = tob(val);
        else      outf[idx] = res[idx] + val;
      }
    }
  }
}

// ---------------------------------------------------------------------------
// Causal flash attention. qkv layout: [N, 3D] rows, q|k|v at col 0|D|2D,
// head h occupies HD=64 columns. 8 waves/block, 16 queries/wave, 32-key chunks.
// ---------------------------------------------------------------------------
__global__ void k_attn(const __bf16* __restrict__ QKV, __bf16* __restrict__ O) {
  constexpr int LD = cfg::D3;
  const int lane = threadIdx.x & 31;
  const int kh   = lane >> 4;
  const int ln16 = lane & 15;
  const int w    = threadIdx.x >> 5;
  const int bh = blockIdx.y;
  const int b  = bh >> 4;
  const int h  = bh & 15;
  const int q0 = blockIdx.x * 128 + w * 16;
  const size_t tokBase = (size_t)b * cfg::S;

  __shared__ __bf16 pb[8 * 16 * 32];     // per-wave P-tile staging (8KB)
  __bf16* pw = pb + w * 512;

  const __bf16* qptr = QKV + (tokBase + q0 + ln16) * LD + h * cfg::HD;
  v16bf qf[2];
  qf[0] = loadA(qptr, 0, kh);
  qf[1] = loadA(qptr, 32, kh);

  v8f of[4];
#pragma unroll
  for (int f = 0; f < 4; ++f) of[f] = zero8();
  float mrun[8], lrun[8];
#pragma unroll
  for (int v = 0; v < 8; ++v) { mrun[v] = -1e30f; lrun[v] = 0.f; }

  const int kend = q0 + 16;              // causal upper bound (exclusive)
  for (int kc = 0; kc < kend; kc += 32) {
    // S = Q K^T (two 16x16 key tiles, K-dim = HD = 64 -> 2 wmma each)
    v8f s[2];
#pragma unroll
    for (int t = 0; t < 2; ++t) {
      s[t] = zero8();
      const __bf16* kptr =
          QKV + (tokBase + kc + t * 16 + ln16) * LD + cfg::D + h * cfg::HD;
      v16bf b0 = loadB(kptr, 0, kh);
      v16bf b1 = loadB(kptr, 32, kh);
      s[t] = wmma_bf16(qf[0], b0, s[t]);
      s[t] = wmma_bf16(qf[1], b1, s[t]);
    }

    // scale + causal mask + per-row max (row lives in one 16-lane half)
    float sv[2][8], rmax[8];
#pragma unroll
    for (int v = 0; v < 8; ++v) {
      const int qrow = q0 + v + 8 * kh;
      float m = -1e30f;
#pragma unroll
      for (int t = 0; t < 2; ++t) {
        const int key = kc + t * 16 + ln16;
        float xv = s[t][v] * 0.125f;      // 1/sqrt(HD)
        if (key > qrow) xv = -1e30f;
        sv[t][v] = xv;
        m = fmaxf(m, xv);
      }
#pragma unroll
      for (int off = 8; off >= 1; off >>= 1) m = fmaxf(m, __shfl_xor(m, off, 32));
      rmax[v] = m;
    }

    // online softmax update; write P (bf16) to LDS for A-layout reload
#pragma unroll
    for (int v = 0; v < 8; ++v) {
      const float mn   = fmaxf(mrun[v], rmax[v]);
      const float corr = __expf(mrun[v] - mn);
      mrun[v] = mn;
      const float p0 = __expf(sv[0][v] - mn);
      const float p1 = __expf(sv[1][v] - mn);
      float rs = p0 + p1;
#pragma unroll
      for (int off = 8; off >= 1; off >>= 1) rs += __shfl_xor(rs, off, 32);
      lrun[v] = lrun[v] * corr + rs;
#pragma unroll
      for (int f = 0; f < 4; ++f) of[f][v] *= corr;
      pw[(v + 8 * kh) * 32 + ln16]      = tob(p0);
      pw[(v + 8 * kh) * 32 + 16 + ln16] = tob(p1);
    }

    // O += P V  (K-dim = 32 keys, 4 HD tiles)
    v16bf pf = loadA(pw + ln16 * 32, 0, kh);
#pragma unroll
    for (int f = 0; f < 4; ++f) {
      v16bf vf;
#pragma unroll
      for (int e = 0; e < 16; ++e) {
        const int key = kc + kh * 16 + e;
        vf[e] = QKV[(tokBase + key) * LD + 2 * cfg::D + h * cfg::HD + f * 16 + ln16];
      }
      of[f] = wmma_bf16(pf, vf, of[f]);
    }
  }

  // normalize and write attn output (bf16, [N, D] with head offset)
#pragma unroll
  for (int f = 0; f < 4; ++f) {
#pragma unroll
    for (int v = 0; v < 8; ++v) {
      const int qrow = q0 + v + 8 * kh;
      const float val = of[f][v] / lrun[v];
      O[(tokBase + qrow) * cfg::D + h * cfg::HD + f * 16 + ln16] = tob(val);
    }
  }
}

// ---------------------------------------------------------------------------
// Fused MoIE FFN1: dual-accumulator GEMM sharing A-fragments:
//   dots = X keys^T, comp = X mu^T (+mu_bias)
// epilogue: scores = dots/(xn*kn); w = relu(scores-gate); masked = comp*w;
//           act = silu(masked) -> bf16
// ---------------------------------------------------------------------------
__global__ void k_ffn1(const __bf16* __restrict__ X,
                       const __bf16* __restrict__ KB,
                       const __bf16* __restrict__ MB,
                       const float* __restrict__ xn,
                       const float* __restrict__ kn,
                       const float* __restrict__ gate,
                       const float* __restrict__ mub,
                       float* __restrict__ scores,
                       float* __restrict__ masked,
                       __bf16* __restrict__ act) {
  const int lane = threadIdx.x & 31;
  const int kh   = lane >> 4;
  const int ln16 = lane & 15;
  const int gtile = blockIdx.x * (blockDim.x >> 5) + (threadIdx.x >> 5);
  const int tilesN = cfg::DFF >> 5;
  const int tm = gtile / tilesN, tn = gtile % tilesN;
  const int r0 = tm * 32, c0 = tn * 32;

  const __bf16* xr0 = X + (size_t)(r0 + ln16) * cfg::D;
  const __bf16* xr1 = xr0 + (size_t)16 * cfg::D;
  const __bf16* kc0 = KB + (size_t)(c0 + ln16) * cfg::D;
  const __bf16* kc1 = kc0 + (size_t)16 * cfg::D;
  const __bf16* mc0 = MB + (size_t)(c0 + ln16) * cfg::D;
  const __bf16* mc1 = mc0 + (size_t)16 * cfg::D;

  v8f accD[2][2], accC[2][2];
#pragma unroll
  for (int i = 0; i < 2; ++i)
#pragma unroll
    for (int j = 0; j < 2; ++j) { accD[i][j] = zero8(); accC[i][j] = zero8(); }

  for (int k0 = 0; k0 < cfg::D; k0 += 32) {
    v16bf a0 = loadA(xr0, k0, kh);
    v16bf a1 = loadA(xr1, k0, kh);
    v16bf bk0 = loadB(kc0, k0, kh);
    v16bf bk1 = loadB(kc1, k0, kh);
    v16bf bm0 = loadB(mc0, k0, kh);
    v16bf bm1 = loadB(mc1, k0, kh);
    accD[0][0] = wmma_bf16(a0, bk0, accD[0][0]);
    accD[0][1] = wmma_bf16(a0, bk1, accD[0][1]);
    accD[1][0] = wmma_bf16(a1, bk0, accD[1][0]);
    accD[1][1] = wmma_bf16(a1, bk1, accD[1][1]);
    accC[0][0] = wmma_bf16(a0, bm0, accC[0][0]);
    accC[0][1] = wmma_bf16(a0, bm1, accC[0][1]);
    accC[1][0] = wmma_bf16(a1, bm0, accC[1][0]);
    accC[1][1] = wmma_bf16(a1, bm1, accC[1][1]);
  }

#pragma unroll
  for (int i = 0; i < 2; ++i) {
#pragma unroll
    for (int j = 0; j < 2; ++j) {
      const int col = c0 + 16 * j + ln16;
      const float knc = kn[col], gc = gate[col], mbc = mub[col];
#pragma unroll
      for (int v = 0; v < 8; ++v) {
        const int row = r0 + 16 * i + v + 8 * kh;
        const size_t idx = (size_t)row * cfg::DFF + col;
        const float sc = accD[i][j][v] / (xn[row] * knc);
        const float wg = fmaxf(sc - gc, 0.f);
        const float mk = (accC[i][j][v] + mbc) * wg;
        scores[idx] = sc;
        masked[idx] = mk;
        act[idx] = tob(mk / (1.f + __expf(-mk)));   // silu
      }
    }
  }
}

// ---------------------------------------------------------------------------
// launch
// ---------------------------------------------------------------------------
extern "C" void kernel_launch(void* const* d_in, const int* in_sizes, int n_in,
                              void* d_out, int out_size, void* d_ws, size_t ws_size,
                              hipStream_t stream) {
  using namespace cfg;
  (void)in_sizes; (void)n_in; (void)out_size; (void)ws_size;

  const float* x      = (const float*)d_in[0];
  const float* ln1_w  = (const float*)d_in[1];
  const float* ln1_b  = (const float*)d_in[2];
  const float* in_w   = (const float*)d_in[3];
  const float* in_b   = (const float*)d_in[4];
  const float* out_w  = (const float*)d_in[5];
  const float* out_b  = (const float*)d_in[6];
  const float* ln2_w  = (const float*)d_in[7];
  const float* ln2_b  = (const float*)d_in[8];
  const float* mu_w   = (const float*)d_in[9];
  const float* sig_w  = (const float*)d_in[10];
  const float* gate   = (const float*)d_in[11];
  const float* mu_b   = (const float*)d_in[12];
  const float* ffn2_w = (const float*)d_in[13];
  const float* ffn2_b = (const float*)d_in[14];

  float* out_x      = (float*)d_out;                    // [N,D]
  float* out_scores = out_x + (size_t)N * D;            // [N,DFF]
  float* out_masked = out_scores + (size_t)N * DFF;     // [N,DFF]

  char* ws = (char*)d_ws;
  __bf16* h_bf    = (__bf16*)(ws + 0);                          // 16MB
  __bf16* qkv_bf  = (__bf16*)(ws + ((size_t)16 << 20));         // 48MB
  __bf16* act_bf  = (__bf16*)(ws + 0);                          // 64MB, aliases h+qkv (dead by FFN1)
  __bf16* attn_bf = (__bf16*)(ws + ((size_t)64 << 20));         // 16MB
  __bf16* ffin_bf = (__bf16*)(ws + ((size_t)80 << 20));         // 16MB
  __bf16* wqkv_bf = (__bf16*)(ws + ((size_t)96 << 20));         // 6MB
  __bf16* wout_bf = (__bf16*)(ws + ((size_t)102 << 20));        // 2MB
  __bf16* wff2_bf = (__bf16*)(ws + ((size_t)104 << 20));        // 8MB
  __bf16* keys_bf = (__bf16*)(ws + ((size_t)112 << 20));        // 8MB
  __bf16* mu_bf   = (__bf16*)(ws + ((size_t)120 << 20));        // 8MB
  float*  xn      = (float*)(ws + ((size_t)128 << 20));         // 32KB
  float*  kn      = (float*)(ws + ((size_t)128 << 20) + (64 << 10)); // 16KB

  // 1) weights -> bf16, keys + norms
  k_cvt_bf16<<<2048, 256, 0, stream>>>(in_w,   wqkv_bf, D3 * D);
  k_cvt_bf16<<<1024, 256, 0, stream>>>(out_w,  wout_bf, D * D);
  k_cvt_bf16<<<2048, 256, 0, stream>>>(ffn2_w, wff2_bf, D * DFF);
  k_keys<<<DFF, 256, 0, stream>>>(mu_w, sig_w, keys_bf, mu_bf, kn);

  // 2) LN1 -> h (bf16)
  k_layernorm<<<N, 256, 0, stream>>>(x, ln1_w, ln1_b, h_bf, nullptr);

  // 3) QKV = h @ in_proj_w^T + b   [8192,3072]
  k_gemm_bf16<<<(N / 32) * (D3 / 32) / 8, 256, 0, stream>>>(
      h_bf, wqkv_bf, in_b, nullptr, nullptr, qkv_bf, N, D3, D);

  // 4) causal flash attention -> attn (bf16)
  k_attn<<<dim3(S / 128, B * H), 256, 0, stream>>>(qkv_bf, attn_bf);

  // 5) x = x + attn @ out_proj_w^T + b  -> d_out.x (fp32 residual)
  k_gemm_bf16<<<(N / 32) * (D / 32) / 8, 256, 0, stream>>>(
      attn_bf, wout_bf, out_b, x, out_x, nullptr, N, D, D);

  // 6) LN2 -> ffn_in (bf16) + row norms xn
  k_layernorm<<<N, 256, 0, stream>>>(out_x, ln2_w, ln2_b, ffin_bf, xn);

  // 7) fused MoIE FFN1: scores/masked (d_out) + silu act (bf16)
  k_ffn1<<<(N / 32) * (DFF / 32) / 8, 256, 0, stream>>>(
      ffin_bf, keys_bf, mu_bf, xn, kn, gate, mu_b, out_scores, out_masked, act_bf);

  // 8) x += act @ ffn2_w^T + b (in-place residual on d_out.x)
  k_gemm_bf16<<<(N / 32) * (D / 32) / 8, 256, 0, stream>>>(
      act_bf, wff2_bf, ffn2_b, out_x, out_x, nullptr, N, D, DFF);
}
